// LiquidS4Layer_63685775065305
// MI455X (gfx1250) — matrix operand order repaired and checked by reading, not compiled, not run on hardware
//
#include <hip/hip_runtime.h>
#include <hip/hip_bf16.h>

// Problem constants (from reference): N=64 state, L=4096 seq, BH=4096 rows.
#define N_SSM 64
#define LSEQ  4096
#define BHROW 4096

typedef __attribute__((ext_vector_type(16))) _Float16 v16h;
typedef __attribute__((ext_vector_type(4)))  _Float16 v4h;
typedef __attribute__((ext_vector_type(8)))  float    v8f;

union Frag16 { v16h v; unsigned int u[8]; };

// ---------------------------------------------------------------------------
// Kernel 1: evaluate SSM generating function at the L roots of unity.
// ---------------------------------------------------------------------------
__global__ void s4_at_roots(const float* __restrict__ Lre, const float* __restrict__ Lim,
                            const float* __restrict__ Pre, const float* __restrict__ Pim,
                            const float* __restrict__ Bre, const float* __restrict__ Bim,
                            const float* __restrict__ Cri, const float* __restrict__ log_step,
                            float* __restrict__ ar_re, float* __restrict__ ar_im) {
  const int k = blockIdx.x * blockDim.x + threadIdx.x;   // 0..LSEQ-1
  const float step = __expf(log_step[0]);
  const float theta = 6.283185307179586f * ((float)k / (float)LSEQ);
  float sth, cth;
  __sincosf(theta, &sth, &cth);
  const float om_re = cth, om_im = -sth;                 // Omega = e^{-i theta}
  const float den_re = 1.0f + om_re, den_im = om_im;     // 1 + Omega
  const float inv_dn = 1.0f / (den_re * den_re + den_im * den_im);
  const float num_re = 1.0f - om_re, num_im = -om_im;    // 1 - Omega
  const float s2 = 2.0f / step;
  const float g_re = s2 * (num_re * den_re + num_im * den_im) * inv_dn;
  const float g_im = s2 * (num_im * den_re - num_re * den_im) * inv_dn;
  const float c_re =  2.0f * den_re * inv_dn;            // c = 2/(1+Omega)
  const float c_im = -2.0f * den_im * inv_dn;

  float k00r = 0.f, k00i = 0.f, k01r = 0.f, k01i = 0.f;
  float k10r = 0.f, k10i = 0.f, k11r = 0.f, k11i = 0.f;
  for (int n = 0; n < N_SSM; ++n) {
    const float lr = Lre[n], li = Lim[n];
    const float dr = g_re - lr, di = g_im - li;
    const float idn = 1.0f / (dr * dr + di * di);
    const float ir = dr * idn, ii = -di * idn;           // 1/(g - Lambda)
    const float pr = Pre[n], pi = Pim[n];
    const float br = Bre[n], bi = Bim[n];
    const float cr = Cri[2 * n], ci = Cri[2 * n + 1];
    const float t00r = cr * br + ci * bi, t00i = cr * bi - ci * br;   // conj(C)*B
    const float t01r = cr * pr + ci * pi, t01i = cr * pi - ci * pr;   // conj(C)*P
    const float t10r = pr * br + pi * bi, t10i = pr * bi - pi * br;   // conj(P)*B
    const float t11r = pr * pr + pi * pi;                              // conj(P)*P
    k00r += t00r * ir - t00i * ii;  k00i += t00r * ii + t00i * ir;
    k01r += t01r * ir - t01i * ii;  k01i += t01r * ii + t01i * ir;
    k10r += t10r * ir - t10i * ii;  k10i += t10r * ii + t10i * ir;
    k11r += t11r * ir;              k11i += t11r * ii;
  }
  const float d1r = 1.0f + k11r, d1i = k11i;
  const float idn1 = 1.0f / (d1r * d1r + d1i * d1i);
  const float qr = k01r * k10r - k01i * k10i;
  const float qi = k01r * k10i + k01i * k10r;
  const float sr = (qr * d1r + qi * d1i) * idn1;
  const float si = (qi * d1r - qr * d1i) * idn1;
  const float tr = k00r - sr, ti = k00i - si;
  ar_re[k] = c_re * tr - c_im * ti;
  ar_im[k] = c_re * ti + c_im * tr;
}

// ---------------------------------------------------------------------------
// Kernel 2: K[n] = (1/L) * Re( sum_k at_roots[k] * e^{+2*pi*i*k*n/L} )
// ---------------------------------------------------------------------------
__global__ void s4_idft(const float* __restrict__ ar_re, const float* __restrict__ ar_im,
                        float* __restrict__ Kf, _Float16* __restrict__ Kh) {
  const int n = blockIdx.x * blockDim.x + threadIdx.x;
  const float w = 6.283185307179586f / (float)LSEQ;
  float acc = 0.0f;
  int p = 0;                                             // (k*n) mod L, exact
  for (int k = 0; k < LSEQ; ++k) {
    float s, c;
    __sincosf(w * (float)p, &s, &c);
    acc += ar_re[k] * c - ar_im[k] * s;
    p += n;
    if (p >= LSEQ) p -= LSEQ;
  }
  const float kv = acc * (1.0f / (float)LSEQ);
  Kf[n] = kv;
  Kh[n] = (_Float16)kv;
}

// ---------------------------------------------------------------------------
// Kernel 3: u (f32) -> uh (f16), one pass; amortized over ~16-32 tile reuses.
// ---------------------------------------------------------------------------
__global__ void s4_cvt_u(const float* __restrict__ u, _Float16* __restrict__ uh) {
  const int idx = blockIdx.x * blockDim.x + threadIdx.x;   // per float4
  const float4 f = ((const float4*)u)[idx];
  v4h h;
  h.x = (_Float16)f.x; h.y = (_Float16)f.y; h.z = (_Float16)f.z; h.w = (_Float16)f.w;
  ((v4h*)uh)[idx] = h;
}

// ---------------------------------------------------------------------------
// Kernel 4: the 32 distinct Toeplitz tiles (one per diagonal d = ct-st),
// stored transposed [n][k]: Bt[d][n][k] = K[d*128 + n - k] (causal zero-fill).
// 1 MB total -> L2-resident, shared by all 1024 GEMM workgroups.
// ---------------------------------------------------------------------------
__global__ void s4_build_btiles(const _Float16* __restrict__ Kh, _Float16* __restrict__ Bt) {
  const int idx = blockIdx.x * blockDim.x + threadIdx.x;   // 32*128*128
  const int d = idx >> 14;
  const int n = (idx >> 7) & 127;
  const int k = idx & 127;
  const int t = (d << 7) + n - k;                          // in [-127, 4095]
  Bt[idx] = (t >= 0) ? Kh[t] : (_Float16)0.0f;
}

// ---------------------------------------------------------------------------
// Kernel 5: Y = U @ T + D*U via f16 WMMA, double-buffered async-LDS pipeline.
// 128x128 tile/WG, 8 waves (2x4), each wave 64x32 = 4x2 v_wmma_f32_16x16x32_f16.
// Tiles staged with GLOBAL_LOAD_ASYNC_TO_LDS_B128 (ASYNCcnt), next stage
// issued before computing current (s_wait_asynccnt 16 => current complete).
// ---------------------------------------------------------------------------
#define TM 128
#define TN 128
#define TK 128
#define LDK (TK + 8)   // 272-byte row stride (17 banks): conflict-free frags

__global__ void __launch_bounds__(256)
s4_toeplitz_gemm(const float* __restrict__ u, const _Float16* __restrict__ uh,
                 const _Float16* __restrict__ Bt, const float* __restrict__ Dp,
                 float* __restrict__ y) {
  __shared__ __align__(16) _Float16 As[2][TM][LDK];  // u tile, row-major [m][k]
  __shared__ __align__(16) _Float16 Bs[2][TN][LDK];  // Toeplitz tile, K-major [n][k]

  const int ct  = blockIdx.x;        // output column tile (time t)
  const int bm  = blockIdx.y;        // batch-row tile
  const int tid = threadIdx.x;
  const int lane = tid & 31;
  const int wave = tid >> 5;
  const int wr = wave & 1;           // 2 row-groups of 64
  const int wc = wave >> 1;          // 4 col-groups of 32
  const int n0 = ct * TN;
  const int m0 = bm * TM;

  const unsigned long long uh_base = (unsigned long long)uh;
  const unsigned long long bt_base = (unsigned long long)Bt;
  const unsigned ldsA0 = (unsigned)(size_t)(&As[0][0][0]);
  const unsigned ldsB0 = (unsigned)(size_t)(&Bs[0][0][0]);
  const unsigned bufA = (unsigned)(TM * LDK * 2);    // bytes per A buffer
  const unsigned bufB = (unsigned)(TN * LDK * 2);

  // Per-stage issue: 8 async b128 for A + 8 for B per thread (16-byte lanes).
  auto issue_tiles = [&](int st, int b) {
    const int k0 = st * TK;
    const unsigned ldsa = ldsA0 + (unsigned)b * bufA;
    for (int i = 0; i < 8; ++i) {
      const int e = i * 256 + tid;                    // 0..2047
      const int r = e >> 4, c8 = e & 15;
      const unsigned lds  = ldsa + (unsigned)(r * LDK + (c8 << 3)) * 2u;
      const unsigned voff = (unsigned)((m0 + r) * LSEQ + k0 + (c8 << 3)) * 2u;
      asm volatile("global_load_async_to_lds_b128 %0, %1, %2"
                   :: "v"(lds), "v"(voff), "s"(uh_base) : "memory");
    }
    const unsigned ldsb = ldsB0 + (unsigned)b * bufB;
    const unsigned tbase = (unsigned)((ct - st) * TN * TK) * 2u;  // diagonal tile
    for (int i = 0; i < 8; ++i) {
      const int e = i * 256 + tid;
      const int n = e >> 4, k8 = e & 15;
      const unsigned lds  = ldsb + (unsigned)(n * LDK + (k8 << 3)) * 2u;
      const unsigned voff = tbase + (unsigned)(n * TK + (k8 << 3)) * 2u;
      asm volatile("global_load_async_to_lds_b128 %0, %1, %2"
                   :: "v"(lds), "v"(voff), "s"(bt_base) : "memory");
    }
  };

  v8f acc[4][2] = {};

  issue_tiles(0, 0);                               // prologue
  for (int st = 0; st <= ct; ++st) {
    const int cur = st & 1;
    const bool have_next = (st + 1 <= ct);
    if (have_next) {
      issue_tiles(st + 1, cur ^ 1);                // overlap next fill w/ compute
      asm volatile("s_wait_asynccnt 0x10" ::: "memory");  // current 16 done
    } else {
      asm volatile("s_wait_asynccnt 0x0" ::: "memory");
    }
    __syncthreads();                               // tile visible to all waves

    for (int ks = 0; ks < 4; ++ks) {
      const int kb = ks << 5;
      Frag16 af[4], bf[2];
      // A frags: wave32 16-bit 16x32 layout. lanes 0-15: K = kb+{0,2,4,6,16,18,20,22}; hi lanes +8.
      {
        const int mlo  = (wr << 6) + (lane & 15);
        const int ksel = kb + ((lane >> 4) << 3);
        for (int i = 0; i < 4; ++i) {
          const int m = mlo + (i << 4);
          af[i].u[0] = *(const unsigned int*)&As[cur][m][ksel + 0];
          af[i].u[1] = *(const unsigned int*)&As[cur][m][ksel + 2];
          af[i].u[2] = *(const unsigned int*)&As[cur][m][ksel + 4];
          af[i].u[3] = *(const unsigned int*)&As[cur][m][ksel + 6];
          af[i].u[4] = *(const unsigned int*)&As[cur][m][ksel + 16];
          af[i].u[5] = *(const unsigned int*)&As[cur][m][ksel + 18];
          af[i].u[6] = *(const unsigned int*)&As[cur][m][ksel + 20];
          af[i].u[7] = *(const unsigned int*)&As[cur][m][ksel + 22];
        }
      }
      // B frags: n = lane&15; same K order as A: lanes 0-15 K=kb+{0..7,16..23}; hi lanes +8.
      {
        const int nb   = (wc << 5) + (lane & 15);
        const int ksel = kb + ((lane >> 4) << 3);
        for (int j = 0; j < 2; ++j) {
          const int n = nb + (j << 4);
          for (int v = 0; v < 8; ++v)
            bf[j].u[v] = *(const unsigned int*)&Bs[cur][n][ksel + ((v & 3) << 1) + ((v >> 2) << 4)];
        }
      }
      for (int i = 0; i < 4; ++i)
        for (int j = 0; j < 2; ++j)
          acc[i][j] = __builtin_amdgcn_wmma_f32_16x16x32_f16(
              false, af[i].v, false, bf[j].v, (short)0, acc[i][j], false, false);
    }
    __syncthreads();                               // buffer free for st+2 issue
  }

  // Epilogue: y = acc + D*u. C layout: VGPR r -> M=r (+8 for hi lanes), N=lane&15.
  const float Dv = Dp[0];
  const int colbase = n0 + (wc << 5) + (lane & 15);
  const int rowoff  = (lane >> 4) << 3;
  for (int i = 0; i < 4; ++i) {
    for (int j = 0; j < 2; ++j) {
      const int col = colbase + (j << 4);
      for (int r = 0; r < 8; ++r) {
        const int row = m0 + (wr << 6) + (i << 4) + rowoff + r;
        const size_t off = (size_t)row * LSEQ + col;
        y[off] = acc[i][j][r] + Dv * u[off];
      }
    }
  }
}

// ---------------------------------------------------------------------------
extern "C" void kernel_launch(void* const* d_in, const int* in_sizes, int n_in,
                              void* d_out, int out_size, void* d_ws, size_t ws_size,
                              hipStream_t stream) {
  (void)in_sizes; (void)n_in; (void)out_size; (void)ws_size;
  const float* u        = (const float*)d_in[0];
  const float* Lre      = (const float*)d_in[1];
  const float* Lim      = (const float*)d_in[2];
  const float* Pre      = (const float*)d_in[3];
  const float* Pim      = (const float*)d_in[4];
  const float* Bre      = (const float*)d_in[5];
  const float* Bim      = (const float*)d_in[6];
  const float* Cri      = (const float*)d_in[7];
  const float* Dp       = (const float*)d_in[8];
  const float* log_step = (const float*)d_in[9];

  // ws layout: ar_re[L] | ar_im[L] | Kf[L] (f32) | Kh[L] f16 | Bt[32*128*128] f16 | uh[BH*L] f16
  float* ws    = (float*)d_ws;
  float* ar_re = ws;
  float* ar_im = ws + LSEQ;
  float* Kf    = ws + 2 * LSEQ;
  _Float16* Kh = (_Float16*)(ws + 3 * LSEQ);
  _Float16* Bt = Kh + LSEQ;
  _Float16* uh = Bt + 32 * TN * TK;

  s4_at_roots<<<LSEQ / 256, 256, 0, stream>>>(Lre, Lim, Pre, Pim, Bre, Bim,
                                              Cri, log_step, ar_re, ar_im);
  s4_idft<<<LSEQ / 256, 256, 0, stream>>>(ar_re, ar_im, Kf, Kh);
  s4_cvt_u<<<(BHROW * LSEQ / 4) / 256, 256, 0, stream>>>(u, uh);
  s4_build_btiles<<<(32 * TN * TK) / 256, 256, 0, stream>>>(Kh, Bt);

  dim3 grid(LSEQ / TN, BHROW / TM);   // 32 x 32 workgroups
  s4_toeplitz_gemm<<<grid, 256, 0, stream>>>(u, uh, Bt, Dp, (float*)d_out);
}
